// gcn_tree_82540681494983
// MI455X (gfx1250) — matrix operand order repaired and checked by reading, not compiled, not run on hardware
//
#include <hip/hip_runtime.h>
#include <hip/hip_bf16.h>
#include <stdint.h>

// ---------------------------------------------------------------------------
// Problem constants (match reference)
// ---------------------------------------------------------------------------
#define BN    8
#define NN    2048
#define FH    128     // FIN == H == E == 128
#define LOUT  10
#define TDEPTH 3
#define KC    32      // WMMA K step (bf16 16x16x32)
#define NEGBIG (-9e15f)

typedef __attribute__((ext_vector_type(16))) __bf16 v16bf;
typedef __attribute__((ext_vector_type(8)))  float  v8f;

// ---------------------------------------------------------------------------
// Small conversion / elementwise kernels
// ---------------------------------------------------------------------------
__global__ __launch_bounds__(256)
void cvt_f32_bf16(const float* __restrict__ s, __bf16* __restrict__ d, int n) {
  int i = blockIdx.x * 256 + threadIdx.x;
  if (i < n) d[i] = (__bf16)s[i];
}

// batched transpose-convert: s[batch][rows][cols] (f32) -> d[batch][cols][rows] (bf16)
__global__ __launch_bounds__(256)
void cvt_transpose_f32_bf16(const float* __restrict__ s, __bf16* __restrict__ d,
                            int rows, int cols) {
  int i = blockIdx.x * 256 + threadIdx.x;
  if (i >= rows * cols) return;
  size_t base = (size_t)blockIdx.y * rows * cols;
  int r = i / cols, c = i % cols;
  d[base + (size_t)c * rows + r] = (__bf16)s[base + i];
}

// adjp = bf16(adj + I): folds gconv's add_self into the adjacency (exact in bf16)
__global__ __launch_bounds__(256)
void adj_plus_I_bf16(const float* __restrict__ adj, __bf16* __restrict__ adjp) {
  size_t i = (size_t)blockIdx.x * 256 + threadIdx.x;
  if (i >= (size_t)BN * NN * NN) return;
  int j = (int)(i % NN);
  int r = (int)((i / NN) % NN);
  adjp[i] = (__bf16)(adj[i] + (r == j ? 1.0f : 0.0f));
}

// x = x * mask(n < batch_num_nodes[b]); write fp32 + bf16 (both row-major, A-role)
__global__ __launch_bounds__(256)
void mask_emb(const float* __restrict__ t, const int* __restrict__ bnn,
              float* __restrict__ xf, __bf16* __restrict__ xb) {
  size_t i = (size_t)blockIdx.x * 256 + threadIdx.x;
  if (i >= (size_t)BN * NN * FH) return;
  int n = (int)((i / FH) % NN);
  int b = (int)(i / ((size_t)NN * FH));
  float v = (n < bnn[b]) ? t[i] : 0.0f;
  xf[i] = v;
  xb[i] = (__bf16)v;
}

// ---------------------------------------------------------------------------
// WMMA GEMM: C[b] = act(A[b] (MxK, bf16 row-major) @ B[b] (Kx128) + bias)
//   B operand is supplied PRE-TRANSPOSED: Bt[b][n][k] (n=0..127, K-contiguous),
//   so both A and B tiles stream as global_load_b128 -> ds_store_b128 (no
//   scalar transpose scatter in the hot loop; transpose is paid once by the
//   producer instead of 16x by consumer M-blocks).
//   - 256 threads = 8 wave32s; block tile = 128(M) x 128(N)
//   - double-buffered LDS pipeline; one barrier per K-iteration
//   - A/B fragment packing follows CDNA5 ISA 7.12.2 (bf16 16x16x32)
//   - CBT: store bf16 output transposed ([n][m], ld=M) for B-role consumers
// ---------------------------------------------------------------------------
template<bool BIAS, bool RELU, bool CBT>
__global__ __launch_bounds__(256)
void wmma_gemm(const __bf16* __restrict__ A, size_t a_bstride,
               const __bf16* __restrict__ Bt, size_t b_bstride,
               const float* __restrict__ bias,
               float* __restrict__ Cf, __bf16* __restrict__ Cb, size_t c_bstride,
               int M, int Kd) {
  __shared__ alignas(16) __bf16 As[2][128][40];  // As[buf][m][k], pad->bank spread
  __shared__ alignas(16) __bf16 Bs[2][128][40];  // Bs[buf][n][k]

  const int tid  = threadIdx.x;
  const int w    = tid >> 5;          // wave id 0..7
  const int lane = tid & 31;
  const int hi   = lane >> 4;         // half-wave
  const int lo   = lane & 15;
  const int batch = blockIdx.z;
  const int mblk  = blockIdx.x * 128;

  const __bf16* Ab = A  + (size_t)batch * a_bstride;
  const __bf16* Bb = Bt + (size_t)batch * b_bstride;

  v8f zero = {0.f,0.f,0.f,0.f,0.f,0.f,0.f,0.f};
  v8f acc[8];
#pragma unroll
  for (int t = 0; t < 8; ++t) acc[t] = zero;

  const int tr = tid >> 1;           // tile row 0..127 (2 threads/row), A and B alike
  const int tk = (tid & 1) * 16;     // 16 bf16 (32B) per thread

  // ---- prologue: tile 0 -> LDS buffer 0 ----
  {
    const uint4* ga = (const uint4*)(Ab + (size_t)(mblk + tr) * Kd + tk);
    uint4 a0 = ga[0], a1q = ga[1];
    const uint4* gb = (const uint4*)(Bb + (size_t)tr * Kd + tk);
    uint4 b0 = gb[0], b1q = gb[1];
    *(uint4*)&As[0][tr][tk]     = a0;
    *(uint4*)&As[0][tr][tk + 8] = a1q;
    *(uint4*)&Bs[0][tr][tk]     = b0;
    *(uint4*)&Bs[0][tr][tk + 8] = b1q;
  }
  __syncthreads();

  int cur = 0;
  for (int k0 = 0; k0 < Kd; k0 += KC) {
    const bool has_next = (k0 + KC) < Kd;

    // ---- issue next tile's global loads early (overlap with compute) ----
    uint4 a0, a1q, b0, b1q;
    if (has_next) {
      const uint4* ga = (const uint4*)(Ab + (size_t)(mblk + tr) * Kd + (k0 + KC) + tk);
      a0 = ga[0]; a1q = ga[1];
      const uint4* gb = (const uint4*)(Bb + (size_t)tr * Kd + (k0 + KC) + tk);
      b0 = gb[0]; b1q = gb[1];
      if (k0 + 2 * KC < Kd)  // pull tile k+2 toward L2 (global_prefetch path)
        __builtin_prefetch(Ab + (size_t)(mblk + tr) * Kd + (k0 + 2 * KC) + tk, 0, 0);
    }

    // ---- A fragment (16x32): lane lo = row m; K-pair interleave per ISA ----
    union { v16bf v; uint32_t u[8]; } fa;
    const uint32_t* arow = (const uint32_t*)&As[cur][w * 16 + lo][0];
#pragma unroll
    for (int r = 0; r < 8; ++r) {
      int kb = ((r < 4) ? 0 : 16) + hi * 8 + (r & 3) * 2;  // even K base
      fa.u[r] = arow[kb >> 1];
    }

    // ---- preload 8 B fragments (independent ds_load_b128 pairs) ----
    union { v16bf v; uint32_t u[8]; } fb[8];
#pragma unroll
    for (int nt = 0; nt < 8; ++nt) {
      const uint32_t* brow = (const uint32_t*)&Bs[cur][nt * 16 + lo][0];
#pragma unroll
      for (int r = 0; r < 8; ++r) fb[nt].u[r] = brow[((r < 4) ? 0 : 8) + hi * 4 + (r & 3)];  // K = (r<4?0:16) + hi*8 + (r&3)*2
    }

    // ---- 8 WMMAs (shared A, disjoint accumulators) ----
#pragma unroll
    for (int nt = 0; nt < 8; ++nt)
      acc[nt] = __builtin_amdgcn_wmma_f32_16x16x32_bf16(
          false, fa.v, false, fb[nt].v, (short)0, acc[nt], false, false);

    // ---- stage next tile into the other LDS buffer (all b128) ----
    if (has_next) {
      int nxt = cur ^ 1;
      *(uint4*)&As[nxt][tr][tk]     = a0;
      *(uint4*)&As[nxt][tr][tk + 8] = a1q;
      *(uint4*)&Bs[nxt][tr][tk]     = b0;
      *(uint4*)&Bs[nxt][tr][tk + 8] = b1q;
    }
    __syncthreads();
    cur ^= 1;
  }

  // ---- epilogue: C/D layout — lane lo = col, VGPR r -> row r + 8*hi ----
  float*  Cfb = Cf ? Cf + (size_t)batch * c_bstride : nullptr;
  __bf16* Cbb = Cb ? Cb + (size_t)batch * c_bstride : nullptr;
#pragma unroll
  for (int nt = 0; nt < 8; ++nt) {
    int n = nt * 16 + lo;
    float bv = 0.0f;
    if constexpr (BIAS) bv = bias[n];
#pragma unroll
    for (int r = 0; r < 8; ++r) {
      int m = mblk + w * 16 + r + 8 * hi;
      float v = acc[nt][r] + bv;
      if constexpr (RELU) v = fmaxf(v, 0.0f);
      if (Cfb) Cfb[(size_t)m * FH + n] = v;
      if (Cbb) {
        if constexpr (CBT) Cbb[(size_t)n * M + m] = (__bf16)v;  // B-role consumer
        else               Cbb[(size_t)m * FH + n] = (__bf16)v;
      }
    }
  }
}

// ---------------------------------------------------------------------------
// BatchNorm1d(num_nodes), training mode, identity affine:
// per node n, normalize over (batch, feature) = 8*128 = 1024 values.
// Writes the result TRANSPOSED (xbT[b][e][n]) for the next adj-GEMM's B-role.
// ---------------------------------------------------------------------------
__global__ __launch_bounds__(256)
void bn_norm(const float* __restrict__ t, __bf16* __restrict__ xbT) {
  int n = blockIdx.x, tid = threadIdx.x;
  float s = 0.f, s2 = 0.f;
  for (int i = tid; i < BN * FH; i += 256) {
    int b = i >> 7, e = i & 127;
    float v = t[((size_t)b * NN + n) * FH + e];
    s += v; s2 += v * v;
  }
  __shared__ float rs[256], rq[256];
  rs[tid] = s; rq[tid] = s2; __syncthreads();
  for (int st = 128; st > 0; st >>= 1) {
    if (tid < st) { rs[tid] += rs[tid + st]; rq[tid] += rq[tid + st]; }
    __syncthreads();
  }
  float mean = rs[0] * (1.0f / (BN * FH));
  float var  = rq[0] * (1.0f / (BN * FH)) - mean * mean;
  float inv  = rsqrtf(var + 1e-5f);
  for (int i = tid; i < BN * FH; i += 256) {
    int b = i >> 7, e = i & 127;
    float v = t[((size_t)b * NN + n) * FH + e];
    xbT[((size_t)b * FH + e) * NN + n] = (__bf16)((v - mean) * inv);
  }
}

// t1[b,n] = h[b,n,:].a1 ; t2[b,n] = h[b,n,:].a2
__global__ __launch_bounds__(128)
void attn_dots(const float* __restrict__ hf, const float* __restrict__ a1,
               const float* __restrict__ a2, float* __restrict__ t1,
               float* __restrict__ t2) {
  int n = blockIdx.x, b = blockIdx.y, e = threadIdx.x;
  float p = hf[((size_t)b * NN + n) * FH + e];
  __shared__ float r1[128], r2[128];
  r1[e] = p * a1[e]; r2[e] = p * a2[e];
  __syncthreads();
  for (int st = 64; st > 0; st >>= 1) {
    if (e < st) { r1[e] += r1[e + st]; r2[e] += r2[e + st]; }
    __syncthreads();
  }
  if (e == 0) { t1[(size_t)b * NN + n] = r1[0]; t2[(size_t)b * NN + n] = r2[0]; }
}

// Masked LeakyReLU + row softmax; writes bf16 attention probabilities.
// Row of 2048 scores cached in LDS (one global read of the adj row).
__global__ __launch_bounds__(256)
void attn_softmax(const float* __restrict__ adj, const float* __restrict__ t1,
                  const float* __restrict__ t2, __bf16* __restrict__ att,
                  float d1) {
  int n = blockIdx.x, b = blockIdx.y, tid = threadIdx.x;
  const size_t roff = ((size_t)b * NN + n) * NN;
  const float* arow = adj + roff;
  const float* t2b  = t2 + (size_t)b * NN;
  float r1 = t1[(size_t)b * NN + n];
  __shared__ float sc[NN];
  __shared__ float red[256];
  float lmax = -3.4e38f;
  for (int j = tid; j < NN; j += 256) {
    float av = arow[j];
    bool sel = (av == d1) || (av == -1.0f);
    float e = r1 + t2b[j];
    e = (e > 0.f) ? e : 0.2f * e;            // LeakyReLU(0.2)
    float s = sel ? e : NEGBIG;
    sc[j] = s;
    lmax = fmaxf(lmax, s);
  }
  red[tid] = lmax; __syncthreads();
  for (int st = 128; st > 0; st >>= 1) {
    if (tid < st) red[tid] = fmaxf(red[tid], red[tid + st]);
    __syncthreads();
  }
  float mx = red[0];
  __syncthreads();
  float lsum = 0.f;
  for (int j = tid; j < NN; j += 256) {
    float p = __expf(sc[j] - mx);
    sc[j] = p; lsum += p;
  }
  red[tid] = lsum; __syncthreads();
  for (int st = 128; st > 0; st >>= 1) {
    if (tid < st) red[tid] += red[tid + st];
    __syncthreads();
  }
  float inv = 1.0f / red[0];
  for (int j = tid; j < NN; j += 256) att[roff + j] = (__bf16)(sc[j] * inv);
}

// gather c_nodes, sum over K=32, project 128->10
__global__ __launch_bounds__(128)
void pool_proj(const float* __restrict__ xf, const int* __restrict__ cn,
               const float* __restrict__ Wp, const float* __restrict__ bp,
               float* __restrict__ out) {
  int b = blockIdx.x, e = threadIdx.x;
  float s = 0.f;
  for (int k = 0; k < 32; ++k) {
    int node = cn[b * 32 + k];
    s += xf[((size_t)b * NN + node) * FH + e];
  }
  __shared__ float pooled[FH];
  pooled[e] = s; __syncthreads();
  if (e < LOUT) {
    float o = bp[e];
    for (int q = 0; q < FH; ++q) o += pooled[q] * Wp[q * LOUT + e];
    out[b * LOUT + e] = o;
  }
}

// ---------------------------------------------------------------------------
// Host orchestration
// ---------------------------------------------------------------------------
extern "C" void kernel_launch(void* const* d_in, const int* in_sizes, int n_in,
                              void* d_out, int out_size, void* d_ws, size_t ws_size,
                              hipStream_t stream) {
  const float* X   = (const float*)d_in[0];
  const float* adj = (const float*)d_in[1];
  const float* W1  = (const float*)d_in[2];
  const float* b1  = (const float*)d_in[3];
  const float* W2  = (const float*)d_in[4];
  const float* b2  = (const float*)d_in[5];
  const float* W3  = (const float*)d_in[6];
  const float* b3  = (const float*)d_in[7];
  const float* Wa  = (const float*)d_in[8];
  const float* a1  = (const float*)d_in[9];
  const float* a2  = (const float*)d_in[10];
  const float* Wp  = (const float*)d_in[11];
  const float* bp  = (const float*)d_in[12];
  const int*   cn  = (const int*)d_in[13];
  const int*   bnn = (const int*)d_in[14];
  float* out = (float*)d_out;
  (void)in_sizes; (void)n_in; (void)out_size; (void)ws_size;

  char* ws = (char*)d_ws;
  size_t off = 0;
  auto carve = [&](size_t bytes) -> char* {
    char* p = ws + off;
    off += (bytes + 255) & ~(size_t)255;
    return p;
  };
  __bf16* adjp = (__bf16*)carve((size_t)BN * NN * NN * 2);  // reused as att later
  __bf16* xbT  = (__bf16*)carve((size_t)BN * NN * FH * 2);  // B-role x (GCN phase)
  __bf16* xbA  = (__bf16*)carve((size_t)BN * NN * FH * 2);  // A-role x (attention)
  __bf16* yb   = (__bf16*)carve((size_t)BN * NN * FH * 2);  // A-role gconv temp
  __bf16* hbT  = (__bf16*)carve((size_t)BN * NN * FH * 2);  // B-role h (attention)
  __bf16* wbT  = (__bf16*)carve((size_t)6 * FH * FH * 2);   // W1,W2,W3,Wa[0..2] ^T
  float*  tf   = (float*)carve((size_t)BN * NN * FH * 4);
  float*  xf   = (float*)carve((size_t)BN * NN * FH * 4);
  float*  hf   = (float*)carve((size_t)BN * NN * FH * 4);
  float*  t1   = (float*)carve((size_t)BN * NN * 4);
  float*  t2   = (float*)carve((size_t)BN * NN * 4);

  const int WE = FH * FH;
  // weights -> bf16, pre-transposed for the GEMM's B side
  cvt_transpose_f32_bf16<<<dim3((WE + 255) / 256, 1), 256, 0, stream>>>(W1, wbT + 0 * WE, FH, FH);
  cvt_transpose_f32_bf16<<<dim3((WE + 255) / 256, 1), 256, 0, stream>>>(W2, wbT + 1 * WE, FH, FH);
  cvt_transpose_f32_bf16<<<dim3((WE + 255) / 256, 1), 256, 0, stream>>>(W3, wbT + 2 * WE, FH, FH);
  cvt_transpose_f32_bf16<<<dim3((WE + 255) / 256, 3), 256, 0, stream>>>(Wa, wbT + 3 * WE, FH, FH);

  size_t adjN = (size_t)BN * NN * NN;
  adj_plus_I_bf16<<<(unsigned)((adjN + 255) / 256), 256, 0, stream>>>(adj, adjp);
  int xN = BN * NN * FH;
  // X -> bf16, transposed per batch ([N][F] -> [F][N]) for the B side
  cvt_transpose_f32_bf16<<<dim3((NN * FH + 255) / 256, BN), 256, 0, stream>>>(X, xbT, NN, FH);

  dim3 gg(NN / 128, 1, BN);
  const size_t sAdj = (size_t)NN * NN, sX = (size_t)NN * FH;

  // GCN layer 1: relu((adj+I)@x @ W1 + b1) -> BN -> xbT
  wmma_gemm<false, false, false><<<gg, 256, 0, stream>>>(adjp, sAdj, xbT, sX, nullptr, nullptr, yb, sX, NN, NN);
  wmma_gemm<true,  true,  false><<<gg, 256, 0, stream>>>(yb, sX, wbT + 0 * WE, 0, b1, tf, nullptr, sX, NN, FH);
  bn_norm<<<NN, 256, 0, stream>>>(tf, xbT);
  // GCN layer 2
  wmma_gemm<false, false, false><<<gg, 256, 0, stream>>>(adjp, sAdj, xbT, sX, nullptr, nullptr, yb, sX, NN, NN);
  wmma_gemm<true,  true,  false><<<gg, 256, 0, stream>>>(yb, sX, wbT + 1 * WE, 0, b2, tf, nullptr, sX, NN, FH);
  bn_norm<<<NN, 256, 0, stream>>>(tf, xbT);
  // GCN layer 3 (no act/bn) + embedding mask -> row-major xf/xbA (A-role now)
  wmma_gemm<false, false, false><<<gg, 256, 0, stream>>>(adjp, sAdj, xbT, sX, nullptr, nullptr, yb, sX, NN, NN);
  wmma_gemm<true,  false, false><<<gg, 256, 0, stream>>>(yb, sX, wbT + 2 * WE, 0, b3, tf, nullptr, sX, NN, FH);
  mask_emb<<<(xN + 255) / 256, 256, 0, stream>>>(tf, bnn, xf, xbA);

  // Tree-depth attention; att reuses adjp storage (GCN stack is done with it)
  __bf16* att = adjp;
  for (int i = 0; i < TDEPTH; ++i) {
    float d1 = (float)(TDEPTH - i);
    // h = x @ Wa[i]: hf row-major (for dots), hbT transposed (B-role of att@h)
    wmma_gemm<false, false, true ><<<gg, 256, 0, stream>>>(xbA, sX, wbT + (3 + i) * WE, 0, nullptr, hf, hbT, sX, NN, FH);
    attn_dots<<<dim3(NN, BN), 128, 0, stream>>>(hf, a1 + i * FH, a2 + i * FH, t1, t2);
    attn_softmax<<<dim3(NN, BN), 256, 0, stream>>>(adj, t1, t2, att, d1);
    // x = att @ h: xf row-major + xbA row-major (A-role of next depth)
    wmma_gemm<false, false, false><<<gg, 256, 0, stream>>>(att, sAdj, hbT, sX, nullptr, xf, xbA, sX, NN, NN);
  }

  pool_proj<<<BN, 128, 0, stream>>>(xf, cn, Wp, bp, out);
}
